// GCN_20779051778398
// MI455X (gfx1250) — compile-verified, matrix operand order checked by reading, not run on hardware
//
#include <hip/hip_runtime.h>
#include <hip/hip_bf16.h>
#include <stdint.h>

typedef __bf16 bf16;
typedef __attribute__((ext_vector_type(16))) __bf16 v16bf;
typedef __attribute__((ext_vector_type(8)))  __bf16 v8bf;
typedef __attribute__((ext_vector_type(4)))  __bf16 v4bf;
typedef __attribute__((ext_vector_type(8)))  float  v8f;

// ---------- bf16 helpers (bit-exact RNE split, no reliance on __bf16 arithmetic) ----------
__device__ __forceinline__ bf16 f2bf(float f) {
  unsigned u = __float_as_uint(f);
  u += 0x7FFFu + ((u >> 16) & 1u);            // round-to-nearest-even
  unsigned short s = (unsigned short)(u >> 16);
  return __builtin_bit_cast(bf16, s);
}
__device__ __forceinline__ float bf2f(bf16 h) {
  unsigned short s = __builtin_bit_cast(unsigned short, h);
  return __uint_as_float(((unsigned)s) << 16);
}

// Load one 16x32 bf16 WMMA operand fragment from an LDS row (K-major per lane).
// Lane layout per CDNA5 ISA: lanes 0-15 hold K={0..7,16..23}, lanes 16-31 hold K={8..15,24..31}.
__device__ __forceinline__ v16bf ldfrag(const bf16* row, int half) {
  v8bf a = *(const v8bf*)(row + half * 8);        // K = half*8 .. half*8+7
  v8bf b = *(const v8bf*)(row + 16 + half * 8);   // K = 16+half*8 .. +7
  return __builtin_shufflevector(a, b, 0,1,2,3,4,5,6,7,8,9,10,11,12,13,14,15);
}

// ---------- split-bf16 WMMA GEMM: C[M,NCOLS] = A[M,K] * B[K,NCOLS] ----------
template<int K, int NCOLS>
__global__ __launch_bounds__(256, 2)
void gemm_split_kernel(const float* __restrict__ A, const float* __restrict__ B,
                       float* __restrict__ C, int M) {
  constexpr int BM = 128, BN = 64, BK = 32;
  constexpr int LDA = BK + 8; // stride 40 bf16 = 80B: keeps 16B alignment, bank-friendly
  __shared__ bf16 sAhi[BM * LDA];
  __shared__ bf16 sAlo[BM * LDA];
  __shared__ bf16 sBhi[BN * LDA];   // B stored transposed: [n][k]
  __shared__ bf16 sBlo[BN * LDA];

  const int tid  = threadIdx.x;
  const int wave = tid >> 5;
  const int lane = tid & 31;
  const int half = lane >> 4;
  const int ml   = lane & 15;
  const int wm   = wave >> 1;                 // 0..3
  const int wn   = wave & 1;                  // 0..1
  const int bm0  = blockIdx.x * BM;
  const int bn0  = blockIdx.y * BN;

  v8f acc[2][2];
  #pragma unroll
  for (int i = 0; i < 2; ++i)
    #pragma unroll
    for (int j = 0; j < 2; ++j)
      #pragma unroll
      for (int q = 0; q < 8; ++q) acc[i][j][q] = 0.0f;

  for (int k0 = 0; k0 < K; k0 += BK) {
    // ---- stage A tile (128x32 fp32 -> bf16 hi/lo planes) ----
    #pragma unroll
    for (int j = 0; j < 4; ++j) {
      int lin = tid + j * 256;                // 0..1023
      int r   = lin >> 3;                     // row in tile
      int c4  = lin & 7;                      // float4 column
      float4 v = make_float4(0.f, 0.f, 0.f, 0.f);
      int row = bm0 + r;
      if (row < M) v = *(const float4*)(A + (size_t)row * K + k0 + c4 * 4);
      float fs[4] = {v.x, v.y, v.z, v.w};
      v4bf hi, lo;
      #pragma unroll
      for (int q = 0; q < 4; ++q) {
        bf16 h = f2bf(fs[q]);
        hi[q] = h;
        lo[q] = f2bf(fs[q] - bf2f(h));
      }
      *(v4bf*)(sAhi + r * LDA + c4 * 4) = hi;
      *(v4bf*)(sAlo + r * LDA + c4 * 4) = lo;
    }
    // ---- stage B tile (32x64 fp32 -> transposed bf16 hi/lo planes) ----
    #pragma unroll
    for (int j = 0; j < 2; ++j) {
      int lin = tid + j * 256;                // 0..511
      int r   = lin >> 4;                     // k row 0..31
      int c4  = lin & 15;                     // n/4
      float4 v = *(const float4*)(B + (size_t)(k0 + r) * NCOLS + bn0 + c4 * 4);
      float fs[4] = {v.x, v.y, v.z, v.w};
      #pragma unroll
      for (int q = 0; q < 4; ++q) {
        int n = c4 * 4 + q;
        bf16 h = f2bf(fs[q]);
        sBhi[n * LDA + r] = h;
        sBlo[n * LDA + r] = f2bf(fs[q] - bf2f(h));
      }
    }
    __syncthreads();

    v16bf fAhi[2], fAlo[2], fBhi[2], fBlo[2];
    #pragma unroll
    for (int t = 0; t < 2; ++t) {
      int m = wm * 32 + t * 16 + ml;
      fAhi[t] = ldfrag(sAhi + m * LDA, half);
      fAlo[t] = ldfrag(sAlo + m * LDA, half);
      int n = wn * 32 + t * 16 + ml;
      fBhi[t] = ldfrag(sBhi + n * LDA, half);
      fBlo[t] = ldfrag(sBlo + n * LDA, half);
    }
    #pragma unroll
    for (int tm = 0; tm < 2; ++tm)
      #pragma unroll
      for (int tn = 0; tn < 2; ++tn) {
        acc[tm][tn] = __builtin_amdgcn_wmma_f32_16x16x32_bf16(
            false, fAhi[tm], false, fBhi[tn], (short)0, acc[tm][tn], false, false);
        acc[tm][tn] = __builtin_amdgcn_wmma_f32_16x16x32_bf16(
            false, fAhi[tm], false, fBlo[tn], (short)0, acc[tm][tn], false, false);
        acc[tm][tn] = __builtin_amdgcn_wmma_f32_16x16x32_bf16(
            false, fAlo[tm], false, fBhi[tn], (short)0, acc[tm][tn], false, false);
      }
    __syncthreads();
  }

  // ---- write back: D layout = VGPR r, lanes 0-15 -> M=r, lanes 16-31 -> M=r+8 ----
  #pragma unroll
  for (int tm = 0; tm < 2; ++tm)
    #pragma unroll
    for (int tn = 0; tn < 2; ++tn)
      #pragma unroll
      for (int r = 0; r < 8; ++r) {
        int row = bm0 + wm * 32 + tm * 16 + r + half * 8;
        int col = bn0 + wn * 32 + tn * 16 + ml;
        if (row < M) C[(size_t)row * NCOLS + col] = acc[tm][tn][r];
      }
}

// ---------- degree / normalization ----------
__global__ void deg_init_kernel(float* deg, int N) {
  int i = blockIdx.x * blockDim.x + threadIdx.x;
  if (i < N) deg[i] = 1.0f;                    // self-loop contributes 1
}
__global__ void deg_count_kernel(const int* __restrict__ dst, float* deg, int E) {
  int e = blockIdx.x * blockDim.x + threadIdx.x;
  if (e < E) atomicAdd(&deg[dst[e]], 1.0f);
}
__global__ void deg_rsqrt_kernel(float* deg, int N) {
  int i = blockIdx.x * blockDim.x + threadIdx.x;
  if (i < N) deg[i] = rsqrtf(deg[i]);          // deg >= 1 always
}

// ---------- edge scatter: out[dst] += h[src] * dinv[src]*dinv[dst] ----------
template<int C>
__global__ __launch_bounds__(256)
void scatter_kernel(const int* __restrict__ src, const int* __restrict__ dst,
                    const float* __restrict__ dinv, const float* __restrict__ h,
                    float* __restrict__ out, int E) {
  constexpr int TPE = C / 4;                   // threads per edge (float4 each)
  constexpr int EPB = 256 / TPE;               // edges per block
  int eslot = threadIdx.x / TPE;
  int c4    = threadIdx.x % TPE;
  int e     = blockIdx.x * EPB + eslot;
  if (e >= E) return;
  int s = src[e], d = dst[e];
  float nrm = dinv[s] * dinv[d];
  float4 v = *(const float4*)(h + (size_t)s * C + c4 * 4);
  float* op = out + (size_t)d * C + c4 * 4;
  atomicAdd(op + 0, v.x * nrm);
  atomicAdd(op + 1, v.y * nrm);
  atomicAdd(op + 2, v.z * nrm);
  atomicAdd(op + 3, v.w * nrm);
}

// ---------- finalize: out += self-loop term + bias (+ReLU) ----------
template<int C, bool RELU>
__global__ __launch_bounds__(256)
void finalize_kernel(float* __restrict__ out_agg, const float* __restrict__ h,
                     const float* __restrict__ dinv, const float* __restrict__ b, int N) {
  long long i = (long long)blockIdx.x * blockDim.x + threadIdx.x;
  if (i >= (long long)N * C) return;
  int node = (int)(i / C);
  int c    = (int)(i % C);
  float di = dinv[node];
  float v = out_agg[i] + h[i] * di * di + b[c];
  out_agg[i] = RELU ? fmaxf(v, 0.0f) : v;
}

extern "C" void kernel_launch(void* const* d_in, const int* in_sizes, int n_in,
                              void* d_out, int out_size, void* d_ws, size_t ws_size,
                              hipStream_t stream) {
  (void)n_in; (void)ws_size;
  const float* x  = (const float*)d_in[0];
  const int*   ei = (const int*)d_in[1];     // [2, E]: src then dst
  const float* W1 = (const float*)d_in[2];
  const float* b1 = (const float*)d_in[3];
  const float* W2 = (const float*)d_in[4];
  const float* b2 = (const float*)d_in[5];
  float* out = (float*)d_out;

  const int IN = 512, HID = 256, OUT = 64;
  const int N = in_sizes[0] / IN;
  const int E = in_sizes[1] / 2;
  const int* srcv = ei;
  const int* dstv = ei + E;

  char* ws = (char*)d_ws;
  float* h1   = (float*)ws; ws += (size_t)N * HID * sizeof(float);
  float* a1   = (float*)ws; ws += (size_t)N * HID * sizeof(float);
  float* h2   = (float*)ws; ws += (size_t)N * OUT * sizeof(float);
  float* dinv = (float*)ws;

  // 1) degrees -> dinv
  deg_init_kernel<<<(N + 255) / 256, 256, 0, stream>>>(dinv, N);
  deg_count_kernel<<<(E + 255) / 256, 256, 0, stream>>>(dstv, dinv, E);
  deg_rsqrt_kernel<<<(N + 255) / 256, 256, 0, stream>>>(dinv, N);

  // 2) h1 = x @ W1   (split-bf16 WMMA)
  {
    dim3 grid((N + 127) / 128, HID / 64);
    gemm_split_kernel<512, 256><<<grid, 256, 0, stream>>>(x, W1, h1, N);
  }

  // 3) a1 = segment_sum(norm * h1[src]) ; finalize adds self-loop + b1, ReLU
  hipMemsetAsync(a1, 0, (size_t)N * HID * sizeof(float), stream);
  scatter_kernel<256><<<(E + 3) / 4, 256, 0, stream>>>(srcv, dstv, dinv, h1, a1, E);
  {
    long long tot = (long long)N * HID;
    finalize_kernel<256, true><<<(unsigned)((tot + 255) / 256), 256, 0, stream>>>(a1, h1, dinv, b1, N);
  }

  // 4) h2 = a1 @ W2
  {
    dim3 grid((N + 127) / 128, OUT / 64);
    gemm_split_kernel<256, 64><<<grid, 256, 0, stream>>>(a1, W2, h2, N);
  }

  // 5) out = segment_sum(norm * h2[src]) + self-loop + b2
  hipMemsetAsync(out, 0, (size_t)out_size * sizeof(float), stream);
  scatter_kernel<64><<<(E + 15) / 16, 256, 0, stream>>>(srcv, dstv, dinv, h2, out, E);
  {
    long long tot = (long long)N * OUT;
    finalize_kernel<64, false><<<(unsigned)((tot + 255) / 256), 256, 0, stream>>>(out, h2, dinv, b2, N);
  }
}
